// PTB_LSTM_5471788335630
// MI455X (gfx1250) — matrix operand order repaired and checked by reading, not compiled, not run on hardware
//
#include <hip/hip_runtime.h>
#include <hip/hip_bf16.h>

// ---------- types ----------
typedef __attribute__((ext_vector_type(16))) __bf16 v16bf;
typedef __attribute__((ext_vector_type(8)))  __bf16 v8bf;
typedef __attribute__((ext_vector_type(8)))  float  v8f;
typedef __attribute__((ext_vector_type(4)))  unsigned int v4u;
typedef __attribute__((ext_vector_type(8)))  int    v8i;
typedef __attribute__((ext_vector_type(4)))  int    v4i;

#define NTOKEN 10000
#define NINP   800
#define NHID   800
#define FOURH  3200
#define SEQ    256
#define BS     20
#define MROWS  (SEQ*BS)       // 5120

// ---------- helpers ----------
__device__ __forceinline__ unsigned short f2bf_bits(float f) {
  unsigned u = __builtin_bit_cast(unsigned, f);
  unsigned r = u + 0x7FFFu + ((u >> 16) & 1u);   // round-to-nearest-even
  return (unsigned short)(r >> 16);
}
__device__ __forceinline__ v8bf ld8(const unsigned short* p) {
  return __builtin_bit_cast(v8bf, *(const v4u*)p);
}
__device__ __forceinline__ v16bf mk16(v8bf lo, v8bf hi) {
  return __builtin_shufflevector(lo, hi, 0,1,2,3,4,5,6,7,8,9,10,11,12,13,14,15);
}
__device__ __forceinline__ v8f wmma_bf16(v16bf a, v16bf b, v8f c) {
  return __builtin_amdgcn_wmma_f32_16x16x32_bf16(false, a, false, b, (short)0, c, false, false);
}
__device__ __forceinline__ float sigm(float x)      { return 1.0f / (1.0f + __expf(-x)); }
__device__ __forceinline__ float tanh_fast(float x) { return 2.0f / (1.0f + __expf(-2.0f * x)) - 1.0f; }

// ----------------------------------------------------------------------------
// TDM: issue a 2D tile load Global(bf16,row-major)->LDS with per-row padding.
// Tile = tile_rows x tile_k bf16 elements; LDS rows padded 64B data + 16B pad
// (pad_interval=3 -> every 16 DWORDs, pad_amount=3 -> 4 DWORDs), matching
// LROW=40 ushorts. Rows >= valid_rows are zero-filled by TDM OOB handling.
// D# bitfields per CDNA5 ISA 8.3/8.4; 6-arg builtin (g0,g1,g2,g3,g_extra,cpol).
// ----------------------------------------------------------------------------
__device__ __forceinline__ void tdm_load_tile(unsigned lds_off, const unsigned short* gptr,
                                              unsigned tile_rows, unsigned valid_rows,
                                              unsigned tile_k, unsigned row_stride_elems) {
  unsigned long long ga = (unsigned long long)(size_t)gptr;
  v4u g0;
  g0[0] = 1u;                                   // count=1, user mode, no gather
  g0[1] = lds_off;                              // lds_addr (wave-relative byte offset)
  g0[2] = (unsigned)(ga & 0xFFFFFFFFull);       // global_addr[31:0]
  g0[3] = (unsigned)((ga >> 32) & 0x01FFFFFFull) | (2u << 30);  // addr[56:32] | type=2
  v8i g1;
  g1[0] = (int)((1u << 16) | (1u << 20) | (3u << 22) | (3u << 25)); // data_size=2B, pad_en, ivl=64B, amt=16B
  g1[1] = (int)((tile_k & 0xFFFFu) << 16);                          // tensor_dim0[15:0]
  g1[2] = (int)(((tile_k >> 16) & 0xFFFFu) | ((valid_rows & 0xFFFFu) << 16)); // td0[31:16] | td1[15:0]
  g1[3] = (int)(((valid_rows >> 16) & 0xFFFFu) | ((tile_k & 0xFFFFu) << 16)); // td1[31:16] | tile_dim0
  g1[4] = (int)(tile_rows & 0xFFFFu);                               // tile_dim1 | tile_dim2=0
  g1[5] = (int)row_stride_elems;                                    // tensor_dim0_stride[31:0]
  g1[6] = 0;                                                        // stride0[47:32] | stride1[15:0]
  g1[7] = 0;                                                        // stride1[47:16]
  v4i z4 = {0, 0, 0, 0};
  v8i z8 = {0, 0, 0, 0, 0, 0, 0, 0};
  __builtin_amdgcn_tensor_load_to_lds(g0, g1, z4, z4, z8, 0);
}

// ============================================================================
// GEMM: C[M,N] = A[M,K] * W[N,K]^T (+ bias[N]); A,W bf16 row-major, C fp32.
// WG tile 64x256 (8 waves of 32x64), K-step 32. TDM-fed double-buffered LDS.
// M multiple of 64; N guarded (TDM zero-fill + store guard); K multiple of 32.
// ============================================================================
#define TM 64
#define TN 256
#define TK 32
#define LROW 40   // LDS row stride in ushorts: 64B data + 16B TDM pad

__global__ __launch_bounds__(256) void ptb_gemm_bf16(
    const unsigned short* __restrict__ A, const unsigned short* __restrict__ W,
    const float* __restrict__ bias, float* __restrict__ C,
    int M, int N, int K)
{
  __shared__ unsigned short lA[2][TM * LROW];
  __shared__ unsigned short lW[2][TN * LROW];

  const int tid  = threadIdx.x;
  const int lane = tid & 31;
  const int wid  = tid >> 5;
  const int wm   = wid >> 2;        // 0..1
  const int wn   = wid & 3;         // 0..3
  const int m0   = blockIdx.y * TM;
  const int n0   = blockIdx.x * TN;
  const int lan  = lane & 15;
  const int khalf = lane >> 4;
  const unsigned wrows = (unsigned)((N - n0) < TN ? (N - n0) : TN);
  const int nk = K / TK;

  v8f acc[2][4] = {};

  // prologue: TDM-fetch k-tile 0 into buffer 0 (wave 0 only; TDM ignores EXEC,
  // wave-uniform branch keeps it out of other waves' instruction streams)
  if (wid == 0) {
    tdm_load_tile((unsigned)(size_t)&lA[0][0], A + (size_t)m0 * K, TM, TM, TK, (unsigned)K);
    tdm_load_tile((unsigned)(size_t)&lW[0][0], W + (size_t)n0 * K, TN, wrows, TK, (unsigned)K);
  }

  for (int kt = 0; kt < nk; ++kt) {
    const int cur = kt & 1;
    if (wid == 0) {
      if (kt + 1 < nk) {
        const int k1 = (kt + 1) * TK;
        tdm_load_tile((unsigned)(size_t)&lA[cur ^ 1][0], A + (size_t)m0 * K + k1, TM, TM, TK, (unsigned)K);
        tdm_load_tile((unsigned)(size_t)&lW[cur ^ 1][0], W + (size_t)n0 * K + k1, TN, wrows, TK, (unsigned)K);
        __builtin_amdgcn_s_wait_tensorcnt(2);   // current pair done (in-order), next pair in flight
      } else {
        __builtin_amdgcn_s_wait_tensorcnt(0);
      }
    }
    __syncthreads();   // publish current LDS buffer to all waves

    // ---- fragments + WMMA ----
    v16bf af[2], bfr[4];
    #pragma unroll
    for (int i = 0; i < 2; i++) {
      const unsigned short* p = &lA[cur][(wm*32 + i*16 + lan) * LROW + khalf * 8];
      af[i] = mk16(ld8(p), ld8(p + 16));        // A: k = khalf*8+{0..7}, 16+khalf*8+{0..7}
    }
    #pragma unroll
    for (int j = 0; j < 4; j++) {
      const unsigned short* p = &lW[cur][(wn*64 + j*16 + lan) * LROW + khalf * 8];
      bfr[j] = mk16(ld8(p), ld8(p + 16));       // B: k = khalf*8+{0..7}, 16+khalf*8+{0..7}
    }
    #pragma unroll
    for (int i = 0; i < 2; i++)
      #pragma unroll
      for (int j = 0; j < 4; j++)
        acc[i][j] = wmma_bf16(af[i], bfr[j], acc[i][j]);

    __syncthreads();   // all reads done before this buffer is re-filled
  }

  // ---- epilogue: C/D layout: row = base + khalf*8 + r, col = base + lan ----
  #pragma unroll
  for (int i = 0; i < 2; i++) {
    const int row = m0 + wm*32 + i*16 + khalf*8;
    #pragma unroll
    for (int j = 0; j < 4; j++) {
      const int col = n0 + wn*64 + j*16 + lan;
      if (col < N) {
        const float b = bias ? bias[col] : 0.0f;
        #pragma unroll
        for (int r = 0; r < 8; r++)
          C[(size_t)(row + r) * N + col] = acc[i][j][r] + b;
      }
    }
  }
}

// ============================================================================
// Persistent LSTM scan. 25 WGs x 128 threads = 100 waves.
// Wave (mtile 0..1, ncol 0..49) computes gates[i,f,g,o] tiles for batch rows
// mtile*16..+15 and hidden cols ncol*16..+15; c-state lives in registers.
// h double-buffered in bf16 global (L2 resident); grid barrier per step.
// Writes hidden_seq directly in bf16 for the TDM-fed decoder GEMM.
// ============================================================================
#define SCAN_WGS 25

__device__ __forceinline__ void grid_sync(unsigned* cnt, unsigned target) {
  __syncthreads();
  if (threadIdx.x == 0) {
    __threadfence();
    atomicAdd(cnt, 1u);
    while (*(volatile unsigned*)cnt < target) __builtin_amdgcn_s_sleep(2);
    __threadfence();
  }
  __syncthreads();
}

__global__ __launch_bounds__(128) void ptb_lstm_scan(
    const float* __restrict__ xg,             // [5120, 3200] precomputed x-gates
    const unsigned short* __restrict__ U_bf,  // [3200, 800] bf16 recurrent weights
    unsigned short* __restrict__ h_db,        // [2][32][800] bf16 h double-buffer
    unsigned short* __restrict__ hseq_bf,     // [256,20,800] bf16
    float* __restrict__ h_out,                // [20,800]
    float* __restrict__ c_out,                // [20,800]
    unsigned* __restrict__ cnt)
{
  const int tid   = threadIdx.x;
  const int lane  = tid & 31;
  const int wid   = tid >> 5;
  const int gw    = blockIdx.x * 4 + wid;   // 0..99
  const int mtile = gw & 1;
  const int ncol  = gw >> 1;                // 0..49
  const int lan   = lane & 15;
  const int khalf = lane >> 4;
  const int arow  = mtile * 16 + lan;       // A-fragment source row in h buffer
  const int col   = ncol * 16 + lan;        // hidden column this lane owns

  v8f cstate = {};

  for (int t = 0; t < SEQ; ++t) {
    const unsigned short* hb_cur = h_db + (size_t)(t & 1) * (32 * NHID);
    unsigned short*       hb_nxt = h_db + (size_t)((t + 1) & 1) * (32 * NHID);

    v8f acc[4] = {};
    for (int k0 = 0; k0 < NHID; k0 += 32) {
      const unsigned short* ap = hb_cur + (size_t)arow * NHID + k0 + khalf * 8;
      v16bf a = mk16(ld8(ap), ld8(ap + 16));
      #pragma unroll
      for (int g = 0; g < 4; g++) {
        const unsigned short* bp =
            U_bf + (size_t)(g * NHID + ncol * 16 + lan) * NHID + k0 + khalf * 8;
        v16bf b = mk16(ld8(bp), ld8(bp + 16));
        acc[g] = wmma_bf16(a, b, acc[g]);
      }
    }

    #pragma unroll
    for (int r = 0; r < 8; r++) {
      const int brow = mtile * 16 + khalf * 8 + r;
      const bool ok  = brow < BS;
      const size_t xrow = (size_t)(t * BS + (ok ? brow : 0)) * FOURH;
      const float gi = acc[0][r] + (ok ? xg[xrow + 0 * NHID + col] : 0.0f);
      const float gf = acc[1][r] + (ok ? xg[xrow + 1 * NHID + col] : 0.0f);
      const float gg = acc[2][r] + (ok ? xg[xrow + 2 * NHID + col] : 0.0f);
      const float go = acc[3][r] + (ok ? xg[xrow + 3 * NHID + col] : 0.0f);
      const float iv = sigm(gi), fv = sigm(gf), gv = tanh_fast(gg), ov = sigm(go);
      const float cn = fv * cstate[r] + iv * gv;
      cstate[r] = cn;
      const float hv = ov * tanh_fast(cn);
      if (ok) {
        const unsigned short hb = f2bf_bits(hv);
        hb_nxt[(size_t)brow * NHID + col] = hb;
        hseq_bf[(size_t)(t * BS + brow) * NHID + col] = hb;
        if (t == SEQ - 1) {
          h_out[brow * NHID + col] = hv;
          c_out[brow * NHID + col] = cn;
        }
      }
    }
    grid_sync(cnt, (unsigned)SCAN_WGS * (unsigned)(t + 1));
  }
}

// ============================================================================
// Init kernels: one-time fp32 -> bf16 weight/activation conversion (8 elems
// per thread, packed dword stores), h double-buffer + sync counter reset.
// ============================================================================
__global__ void ptb_cvt_bf16(const float* __restrict__ src,
                             unsigned short* __restrict__ dst, int n)
{
  const int i = (blockIdx.x * blockDim.x + threadIdx.x) * 8;
  if (i + 8 <= n) {
    v4u p;
    #pragma unroll
    for (int e = 0; e < 4; e++)
      p[e] = (unsigned)f2bf_bits(src[i + 2*e]) | ((unsigned)f2bf_bits(src[i + 2*e + 1]) << 16);
    *(v4u*)(dst + i) = p;
  } else {
    for (int j = i; j < n; j++) dst[j] = f2bf_bits(src[j]);
  }
}

__global__ void ptb_state_init(unsigned short* __restrict__ h_db, unsigned* __restrict__ cnt)
{
  const int i = blockIdx.x * blockDim.x + threadIdx.x;
  if (i < 2 * 32 * NHID) h_db[i] = 0;
  if (i == 0) *cnt = 0u;
}

// ============================================================================
// launch
// ============================================================================
extern "C" void kernel_launch(void* const* d_in, const int* in_sizes, int n_in,
                              void* d_out, int out_size, void* d_ws, size_t ws_size,
                              hipStream_t stream) {
  (void)in_sizes; (void)n_in; (void)out_size; (void)ws_size;
  const float* emb   = (const float*)d_in[0];  // [256,20,800]
  const float* W_w   = (const float*)d_in[1];  // [3200,800]
  const float* W_b   = (const float*)d_in[2];  // [3200]
  const float* U_w   = (const float*)d_in[3];  // [3200,800]
  const float* dec_w = (const float*)d_in[4];  // [10000,800]
  const float* dec_b = (const float*)d_in[5];  // [10000]

  float* out   = (float*)d_out;                          // decoded [5120,10000]
  float* h_out = out + (size_t)MROWS * NTOKEN;           // [20,800]
  float* c_out = h_out + (size_t)BS * NHID;              // [20,800]

  char* ws = (char*)d_ws;
  unsigned*       cnt     = (unsigned*)ws;                                   // 256 B
  float*          xg      = (float*)(ws + 256);                              // 65,536,000 B
  unsigned short* U_bf    = (unsigned short*)(ws + 65536256ull);             //  5,120,000 B
  unsigned short* h_db    = (unsigned short*)(ws + 70656256ull);             //    102,400 B
  unsigned short* emb_bf  = (unsigned short*)(ws + 70758656ull);             //  8,192,000 B
  unsigned short* Ww_bf   = (unsigned short*)(ws + 78950656ull);             //  5,120,000 B
  unsigned short* decw_bf = (unsigned short*)(ws + 84070656ull);             // 16,000,000 B
  unsigned short* hseq_bf = (unsigned short*)(ws + 100070656ull);            //  8,192,000 B

  // 0) bf16 conversions + state init
  const int n_emb = MROWS * NINP, n_ww = FOURH * NINP, n_uw = FOURH * NHID, n_dw = NTOKEN * NHID;
  ptb_cvt_bf16<<<(n_emb/8 + 255)/256, 256, 0, stream>>>(emb,   emb_bf,  n_emb);
  ptb_cvt_bf16<<<(n_ww /8 + 255)/256, 256, 0, stream>>>(W_w,   Ww_bf,   n_ww);
  ptb_cvt_bf16<<<(n_uw /8 + 255)/256, 256, 0, stream>>>(U_w,   U_bf,    n_uw);
  ptb_cvt_bf16<<<(n_dw /8 + 255)/256, 256, 0, stream>>>(dec_w, decw_bf, n_dw);
  ptb_state_init<<<(2*32*NHID + 255)/256, 256, 0, stream>>>(h_db, cnt);

  // 1) x_gates = emb * W_w^T + W_b : [5120,3200]
  ptb_gemm_bf16<<<dim3((FOURH + TN - 1) / TN, MROWS / TM), 256, 0, stream>>>(
      emb_bf, Ww_bf, W_b, xg, MROWS, FOURH, NINP);

  // 2) recurrent scan (persistent, grid-synced)
  ptb_lstm_scan<<<SCAN_WGS, 128, 0, stream>>>(xg, U_bf, h_db, hseq_bf, h_out, c_out, cnt);

  // 3) decoded = hseq * dec_w^T + dec_b : [5120,10000]
  ptb_gemm_bf16<<<dim3((NTOKEN + TN - 1) / TN, MROWS / TM), 256, 0, stream>>>(
      hseq_bf, decw_bf, dec_b, out, MROWS, NTOKEN, NHID);
}
